// BilinearAttention_25898652795079
// MI455X (gfx1250) — matrix operand order repaired and checked by reading, not compiled, not run on hardware
//
#include <hip/hip_runtime.h>
#include <hip/hip_bf16.h>

typedef __attribute__((ext_vector_type(16))) __bf16 v16bf;
typedef __attribute__((ext_vector_type(8)))  __bf16 v8bf;
typedef __attribute__((ext_vector_type(8)))  float  v8f;

#define B_   4
#define LQ_  4096
#define LK_  4096
#define D_   256
#define SCALE_ 0.0625f          // 1/sqrt(256)
#define NEG_INF_ (-1000000000.0f)

// ---------------------------------------------------------------------------
// CDNA5 async global->LDS copy (per-lane 16B), tracked by ASYNCcnt.
// LDS operand = low 32 bits of the generic pointer (flat aperture rule).
// ---------------------------------------------------------------------------
__device__ __forceinline__ void async_copy_b128(const void* gptr, void* lptr) {
  unsigned ldsoff = (unsigned)(unsigned long long)lptr;
  asm volatile("global_load_async_to_lds_b128 %0, %1, off"
               :: "v"(ldsoff), "v"((unsigned long long)gptr)
               : "memory");
}
__device__ __forceinline__ void wait_async0() {
  asm volatile("s_wait_asynccnt 0" ::: "memory");
}

// ---------------------------------------------------------------------------
// WMMA fragment loaders (CDNA5 16x16x32 bf16 layouts, wave32)
// A (16x32, 16-bit): row = lane&15 ; kb=(lane>>4)*8 ; [0..7]=K kb.., [8..15]=K 16+kb..
// B (32x16, 16-bit): col = lane&15 ; kb=(lane>>4)*8 ; [0..7]=K kb.., [8..15]=K 16+kb..
// C/D (16x16 f32):   M = r + 8*(lane>>4), N = lane&15
// ---------------------------------------------------------------------------
__device__ __forceinline__ v16bf load_a16x32(const __bf16* base, int ld) {
  const int lane = threadIdx.x & 31;
  const int row  = lane & 15;
  const int kb   = (lane >> 4) * 8;
  const __bf16* p = base + row * ld + kb;
  v8bf lo = *(const v8bf*)(p);
  v8bf hi = *(const v8bf*)(p + 16);
  v16bf a;
#pragma unroll
  for (int i = 0; i < 8; ++i) { a[i] = lo[i]; a[8 + i] = hi[i]; }
  return a;
}

__device__ __forceinline__ v16bf load_b32x16(const __bf16* base, int ld) {
  const int lane = threadIdx.x & 31;
  const int n    = lane & 15;
  const int kb   = (lane >> 4) * 8;
  const __bf16* p = base + n * ld + kb;
  v8bf lo = *(const v8bf*)(p);
  v8bf hi = *(const v8bf*)(p + 16);
  v16bf b;
#pragma unroll
  for (int i = 0; i < 8; ++i) { b[i] = lo[i]; b[8 + i] = hi[i]; }
  return b;
}

// ---------------------------------------------------------------------------
// f32 -> bf16 conversion (8 elements/thread)
// ---------------------------------------------------------------------------
__global__ void __launch_bounds__(256)
cvt_f32_bf16_kernel(const float* __restrict__ src, __bf16* __restrict__ dst, int n8) {
  int i = blockIdx.x * blockDim.x + threadIdx.x;
  if (i >= n8) return;
  const float4* s4 = (const float4*)src + (size_t)i * 2;
  float4 a = s4[0];
  float4 b = s4[1];
  v8bf o;
  o[0] = (__bf16)a.x; o[1] = (__bf16)a.y; o[2] = (__bf16)a.z; o[3] = (__bf16)a.w;
  o[4] = (__bf16)b.x; o[5] = (__bf16)b.y; o[6] = (__bf16)b.z; o[7] = (__bf16)b.w;
  *((v8bf*)dst + i) = o;
}

// w (DxD, row-major) -> wT bf16 (transposed): WMMA B-fragments K-contiguous
__global__ void __launch_bounds__(256)
cvt_w_transpose_kernel(const float* __restrict__ w, __bf16* __restrict__ wT) {
  int lin = blockIdx.x * blockDim.x + threadIdx.x;   // D_*D_ threads
  int d = lin >> 8;
  int e = lin & 255;
  wT[e * D_ + d] = (__bf16)w[lin];
}

// v [B][LK][D] f32 -> vT [B][D][LK] bf16, tiled 64x64 transpose through LDS
// (coalesced on both global sides; LDS pitch 66 -> conflict-free)
__global__ void __launch_bounds__(256)
cvt_v_transpose_kernel(const float* __restrict__ v, __bf16* __restrict__ vT) {
  __shared__ __bf16 sT[64 * 66];
  const int tilesD = D_ / 64;                 // 4
  const int tilesK = LK_ / 64;                // 64
  int t  = blockIdx.x;
  int b  = t / (tilesK * tilesD);
  int r  = t % (tilesK * tilesD);
  int k0 = (r / tilesD) * 64;
  int d0 = (r % tilesD) * 64;

  const float* srcB = v + ((size_t)b * LK_ + k0) * D_ + d0;
#pragma unroll
  for (int i = 0; i < 16; ++i) {
    int lin = threadIdx.x + i * 256;          // 4096 elements
    int kk = lin >> 6;
    int dd = lin & 63;
    sT[kk * 66 + dd] = (__bf16)srcB[(size_t)kk * D_ + dd];
  }
  __syncthreads();
  __bf16* dstB = vT + ((size_t)b * D_ + d0) * LK_ + k0;
#pragma unroll
  for (int i = 0; i < 16; ++i) {
    int lin = threadIdx.x + i * 256;
    int dd = lin >> 6;
    int kk = lin & 63;
    dstB[(size_t)dd * LK_ + kk] = sT[kk * 66 + dd];
  }
}

// ---------------------------------------------------------------------------
// qw = q @ w   (M = B*LQ, N = D, K = D)  bf16 WMMA, one 16x16 tile per wave
// ---------------------------------------------------------------------------
__global__ void __launch_bounds__(128)
qw_gemm_kernel(const __bf16* __restrict__ qb, const __bf16* __restrict__ wT,
               __bf16* __restrict__ qwb) {
  const int wave = threadIdx.x >> 5;
  const int lane = threadIdx.x & 31;
  const int tile = blockIdx.x * 4 + wave;
  const int mt = tile >> 4;
  const int nt = tile & 15;

  const __bf16* aBase = qb + (size_t)mt * 16 * D_;
  const __bf16* bBase = wT + (size_t)nt * 16 * D_;

  v8f acc = {};
#pragma unroll
  for (int k0 = 0; k0 < D_; k0 += 32) {
    v16bf a = load_a16x32(aBase + k0, D_);
    v16bf b = load_b32x16(bBase + k0, D_);
    acc = __builtin_amdgcn_wmma_f32_16x16x32_bf16(false, a, false, b,
                                                  (short)0, acc, false, false);
  }

  const int col    = lane & 15;
  const int rowOff = (lane >> 4) * 8;
  __bf16* o = qwb + (size_t)(mt * 16) * D_ + nt * 16 + col;
#pragma unroll
  for (int r = 0; r < 8; ++r)
    o[(size_t)(rowOff + r) * D_] = (__bf16)acc[r];
}

// ---------------------------------------------------------------------------
// Fused attention, 4 waves/block (each wave = 16 q rows), async double-buffered
// LDS staging of k and vT tiles.
// ---------------------------------------------------------------------------
__global__ void __launch_bounds__(128)
attn_kernel(const __bf16* __restrict__ qwb, const __bf16* __restrict__ kbuf,
            const __bf16* __restrict__ vT, const int* __restrict__ klen,
            float* __restrict__ out, float* __restrict__ att) {
  __shared__ __align__(16) __bf16 sK[2][16 * D_];   // 2 x 8 KB : [kk][e]
  __shared__ __align__(16) __bf16 sV[2][D_ * 32];   // 2 x 16 KB: [d][kk]
  __shared__ float sM[64];
  __shared__ float sS[64];

  const int wave = threadIdx.x >> 5;
  const int lane = threadIdx.x & 31;
  const int b    = blockIdx.x / (LQ_ / 64);
  const int qblk = blockIdx.x % (LQ_ / 64);
  const int qbase = qblk * 64 + wave * 16;

  const __bf16* qRow  = qwb  + ((size_t)b * LQ_ + qbase) * D_;
  const __bf16* kBase = kbuf + (size_t)b * LK_ * D_;
  const __bf16* vTB   = vT   + (size_t)b * D_ * LK_;
  const int*    maskB = klen + b * LK_;
  float* attRow = att + ((size_t)b * LQ_ + qbase) * (size_t)LK_;

  // q fragments are loop-invariant: hoist all 8
  v16bf aQ[8];
#pragma unroll
  for (int e = 0; e < 8; ++e) aQ[e] = load_a16x32(qRow + e * 32, D_);

  float m[8], s[8];
#pragma unroll
  for (int r = 0; r < 8; ++r) { m[r] = -3.0e38f; s[r] = 0.0f; }

  const int col    = lane & 15;
  const int rowOff = (lane >> 4) * 8;

  // ---------------- Pass 1: scores + mask + raw att + online stats ---------
  auto stage_k = [&](int kt, int buf) {
    const __bf16* src = kBase + (size_t)kt * 16 * D_;   // 16 contiguous rows
    __bf16* dst = &sK[buf][0];
#pragma unroll
    for (int i = 0; i < 4; ++i) {
      int c = threadIdx.x + i * 128;                    // 512 chunks of 16B
      async_copy_b128(src + c * 8, dst + c * 8);
    }
  };

  stage_k(0, 0);
  wait_async0();
  __syncthreads();

  for (int kt = 0; kt < LK_ / 16; ++kt) {
    if (kt + 1 < LK_ / 16) stage_k(kt + 1, (kt + 1) & 1);   // overlap with compute

    const __bf16* kTile = &sK[kt & 1][0];
    v8f acc = {};
#pragma unroll
    for (int e = 0; e < 8; ++e) {
      v16bf bfr = load_b32x16(kTile + e * 32, D_);
      acc = __builtin_amdgcn_wmma_f32_16x16x32_bf16(false, aQ[e], false, bfr,
                                                    (short)0, acc, false, false);
    }

    const int mi = maskB[kt * 16 + col];
#pragma unroll
    for (int r = 0; r < 8; ++r) {
      float sc = acc[r] * SCALE_;
      sc = (mi == 0) ? NEG_INF_ : sc;
      acc[r] = sc;
      attRow[(size_t)(rowOff + r) * LK_ + kt * 16 + col] = sc;  // raw scores
    }

    // online row max / sumexp across the 16 lanes of each half-wave
#pragma unroll
    for (int r = 0; r < 8; ++r) {
      float tm = acc[r];
#pragma unroll
      for (int off = 1; off < 16; off <<= 1)
        tm = fmaxf(tm, __shfl_xor(tm, off, 32));
      float nm = fmaxf(m[r], tm);
      float p = __expf(acc[r] - nm);
#pragma unroll
      for (int off = 1; off < 16; off <<= 1)
        p += __shfl_xor(p, off, 32);
      s[r] = s[r] * __expf(m[r] - nm) + p;
      m[r] = nm;
    }

    wait_async0();          // our slice of the next tile has landed
    __syncthreads();        // everyone done with current tile + staging visible
  }

  // publish per-row stats (lanes 0-15 hold rows 0-7, lanes 16-31 rows 8-15)
  if (lane == 0) {
#pragma unroll
    for (int r = 0; r < 8; ++r) { sM[wave * 16 + r] = m[r]; sS[wave * 16 + r] = s[r]; }
  } else if (lane == 16) {
#pragma unroll
    for (int r = 0; r < 8; ++r) { sM[wave * 16 + 8 + r] = m[r]; sS[wave * 16 + 8 + r] = s[r]; }
  }
  __syncthreads();

  const float mRow  = sM[wave * 16 + (lane & 15)];
  const float isRow = 1.0f / sS[wave * 16 + (lane & 15)];

  // ---------------- Pass 2: normalize att + out = att @ v -----------------
  v8f o[16];
#pragma unroll
  for (int nt = 0; nt < 16; ++nt) { v8f z = {}; o[nt] = z; }

  const int kbh  = (lane >> 4) * 8;
  const int aRow = lane & 15;

  auto stage_v = [&](int kc, int buf) {
    __bf16* dst = &sV[buf][0];
#pragma unroll
    for (int i = 0; i < 8; ++i) {
      int c = threadIdx.x + i * 128;        // 1024 chunks of 16B
      int d   = c >> 2;
      int kko = (c & 3) * 8;
      async_copy_b128(vTB + (size_t)d * LK_ + kc * 32 + kko,
                      dst + d * 32 + kko);
    }
  };

  stage_v(0, 0);
  wait_async0();
  __syncthreads();

  for (int kc = 0; kc < LK_ / 32; ++kc) {
    if (kc + 1 < LK_ / 32) stage_v(kc + 1, (kc + 1) & 1);

    // read raw scores, normalize, overwrite att with probabilities, bf16 A frag
    float* rp = attRow + (size_t)aRow * LK_ + kc * 32;
    float lo[8], hi[8];
#pragma unroll
    for (int j = 0; j < 8; ++j) { lo[j] = rp[kbh + j]; hi[j] = rp[16 + kbh + j]; }
    v16bf af;
#pragma unroll
    for (int j = 0; j < 8; ++j) {
      lo[j] = __expf(lo[j] - mRow) * isRow;
      hi[j] = __expf(hi[j] - mRow) * isRow;
      rp[kbh + j]      = lo[j];
      rp[16 + kbh + j] = hi[j];
      af[j]     = (__bf16)lo[j];
      af[8 + j] = (__bf16)hi[j];
    }

    const __bf16* vTile = &sV[kc & 1][0];
#pragma unroll
    for (int nt = 0; nt < 16; ++nt) {
      v16bf bfr = load_b32x16(vTile + nt * 16 * 32, 32);
      o[nt] = __builtin_amdgcn_wmma_f32_16x16x32_bf16(false, af, false, bfr,
                                                      (short)0, o[nt], false, false);
    }

    wait_async0();
    __syncthreads();
  }

  // write out tile (16 x 256)
  float* oBase = out + ((size_t)b * LQ_ + qbase) * D_;
#pragma unroll
  for (int nt = 0; nt < 16; ++nt)
#pragma unroll
    for (int r = 0; r < 8; ++r)
      oBase[(size_t)(rowOff + r) * D_ + nt * 16 + col] = o[nt][r];
}

// ---------------------------------------------------------------------------
extern "C" void kernel_launch(void* const* d_in, const int* in_sizes, int n_in,
                              void* d_out, int out_size, void* d_ws, size_t ws_size,
                              hipStream_t stream) {
  const float* q    = (const float*)d_in[0];
  const float* k    = (const float*)d_in[1];
  const float* v    = (const float*)d_in[2];
  const int*   klen = (const int*)d_in[3];
  const float* w    = (const float*)d_in[4];

  float* out = (float*)d_out;                          // [B, LQ, D]
  float* att = out + (size_t)B_ * LQ_ * D_;            // [B, LQ, LK]

  const size_t nQKV = (size_t)B_ * LQ_ * D_;           // == B*LK*D
  char* ws = (char*)d_ws;
  __bf16* qb  = (__bf16*)ws;  ws += nQKV * 2;
  __bf16* kb  = (__bf16*)ws;  ws += nQKV * 2;
  __bf16* vT  = (__bf16*)ws;  ws += nQKV * 2;
  __bf16* wT  = (__bf16*)ws;  ws += (size_t)D_ * D_ * 2;
  __bf16* qwb = (__bf16*)ws;  ws += nQKV * 2;

  const int n8 = (int)(nQKV / 8);
  cvt_f32_bf16_kernel<<<n8 / 256, 256, 0, stream>>>(q, qb, n8);
  cvt_f32_bf16_kernel<<<n8 / 256, 256, 0, stream>>>(k, kb, n8);
  cvt_v_transpose_kernel<<<B_ * (LK_ / 64) * (D_ / 64), 256, 0, stream>>>(v, vT);
  cvt_w_transpose_kernel<<<(D_ * D_) / 256, 256, 0, stream>>>(w, wT);

  qw_gemm_kernel<<<(B_ * LQ_ / 16) * (D_ / 16) / 4, 128, 0, stream>>>(qb, wT, qwb);

  attn_kernel<<<B_ * (LQ_ / 64), 128, 0, stream>>>(qwb, kb, vT, klen, out, att);
}
